// CWTLatentAttention_34299608825935
// MI455X (gfx1250) — matrix operand order repaired and checked by reading, not compiled, not run on hardware
//
#include <hip/hip_runtime.h>
#include <hip/hip_bf16.h>
#include <cstdint>
#include <cstddef>

// ---------------- types ----------------
typedef __attribute__((ext_vector_type(16))) __bf16 v16bf;
typedef __attribute__((ext_vector_type(8)))  __bf16 v8bf;
typedef __attribute__((ext_vector_type(8)))  float  v8f;

union ABFrag { v16bf v; v8bf h[2]; };

__device__ inline v8f wmma_bf16(v16bf a, v16bf b, v8f c) {
    return __builtin_amdgcn_wmma_f32_16x16x32_bf16(false, a, false, b, (short)0, c, false, false);
}

__device__ inline __bf16 f2bf(float f) { return (__bf16)f; }

// CDNA5 async global->LDS copy (tracked by ASYNCcnt). vdst = LDS byte offset,
// vaddr = 64-bit global address. Each wave instruction moves 32 lanes x 16B.
__device__ inline void async_load_b128(unsigned lds_off, const void* gptr) {
    unsigned long long ga = (unsigned long long)(uintptr_t)gptr;
    asm volatile("global_load_async_to_lds_b128 %0, %1, off"
                 :: "v"(lds_off), "v"(ga) : "memory");
}
__device__ inline unsigned lds_off32(const void* p) {
    return (unsigned)(uintptr_t)p;  // low 32 bits of generic addr == LDS offset
}

// Problem constants
#define Bz 4
#define Tz 2048
#define Dz 2048
#define Hz 16
#define DHz 128
#define DLATz 512
#define DHUBz 1024

// ---------------- fp32 -> bf16 convert ----------------
__global__ __launch_bounds__(256) void cvt_bf16_kernel(const float* __restrict__ src,
                                                       __bf16* __restrict__ dst, size_t n) {
    size_t i = (size_t)blockIdx.x * 256 + threadIdx.x;
    if (i < n) dst[i] = f2bf(src[i]);
}

// ---------------- RMSNorm (per row of 1024) -> bf16 ----------------
__global__ __launch_bounds__(256) void rmsnorm_kernel(const float* __restrict__ hub,
                                                      const float* __restrict__ w,
                                                      __bf16* __restrict__ out) {
    int row = blockIdx.x;
    const float* x = hub + (size_t)row * DHUBz;
    float s = 0.f;
    for (int i = threadIdx.x; i < DHUBz; i += 256) { float v = x[i]; s += v * v; }
    for (int off = 16; off; off >>= 1) s += __shfl_xor(s, off, 32);
    __shared__ float red[8];
    __shared__ float rtot;
    int wv = threadIdx.x >> 5, ln = threadIdx.x & 31;
    if (ln == 0) red[wv] = s;
    __syncthreads();
    if (threadIdx.x == 0) {
        float t = 0.f;
        #pragma unroll
        for (int i = 0; i < 8; ++i) t += red[i];
        rtot = rsqrtf(t * (1.0f / DHUBz) + 1e-6f);
    }
    __syncthreads();
    float r = rtot;
    __bf16* o = out + (size_t)row * DHUBz;
    for (int i = threadIdx.x; i < DHUBz; i += 256) o[i] = f2bf(x[i] * r * w[i]);
}

// ---------------- bf16 GEMM: C[M,N] = A[M,K] * W[N,K]^T ----------------
// Block tile 256x128, 8 waves (4x2), wave tile 64x64 (16 WMMA / K-step), K-step 32.
// Double-buffered LDS filled with async global->LDS copies (ASYNCcnt pipelined).
template <bool OUT_BF16>
__global__ __launch_bounds__(256) void gemm_bf16_kernel(const __bf16* __restrict__ A,
                                                        const __bf16* __restrict__ W,
                                                        void* __restrict__ Cout,
                                                        int M, int N, int K) {
    __shared__ __bf16 sA[2][256 * 40];
    __shared__ __bf16 sB[2][128 * 40];
    const int tid  = threadIdx.x;
    const int lane = tid & 31;
    const int wv   = tid >> 5;
    const int wm   = wv >> 1;        // 0..3  -> M offset wm*64
    const int wn   = wv & 1;         // 0..1  -> N offset wn*64
    const int hb   = lane >> 4;      // half-wave
    const int l15  = lane & 15;
    const int m0 = blockIdx.y * 256;
    const int n0 = blockIdx.x * 128;

    v8f acc[4][4];
    #pragma unroll
    for (int mi = 0; mi < 4; ++mi)
        #pragma unroll
        for (int ni = 0; ni < 4; ++ni) acc[mi][ni] = (v8f)(0.0f);

    auto fill = [&](int buf, int k0) {
        #pragma unroll
        for (int j = 0; j < 4; ++j) {               // A: 256x32 = 1024 16B chunks
            int c = tid + 256 * j;
            int row = c >> 2, ko = (c & 3) * 8;
            const __bf16* sa = A + (size_t)(m0 + row) * K + k0 + ko;
            async_load_b128(lds_off32(&sA[buf][row * 40 + ko]), sa);
        }
        #pragma unroll
        for (int j = 0; j < 2; ++j) {               // B: 128x32 = 512 16B chunks
            int c = tid + 256 * j;
            int row = c >> 2, ko = (c & 3) * 8;
            const __bf16* sb = W + (size_t)(n0 + row) * K + k0 + ko;
            async_load_b128(lds_off32(&sB[buf][row * 40 + ko]), sb);
        }
    };

    const int nk = K >> 5;
    fill(0, 0);
    for (int kb = 0; kb < nk; ++kb) {
        const int cur = kb & 1;
        const bool pf = (kb + 1 < nk);
        if (pf) fill(cur ^ 1, (kb + 1) << 5);
        // Async loads complete in order: waiting for <=6 outstanding means the
        // current buffer's 6 loads from last iteration have landed.
        if (pf) asm volatile("s_wait_asynccnt 0x6" ::: "memory");
        else    asm volatile("s_wait_asynccnt 0x0" ::: "memory");
        __syncthreads();

        ABFrag afr[4];
        #pragma unroll
        for (int mi = 0; mi < 4; ++mi) {
            int row = wm * 64 + mi * 16 + l15;
            int ka = hb * 8;
            afr[mi].h[0] = *(v8bf*)&sA[cur][row * 40 + ka];
            afr[mi].h[1] = *(v8bf*)&sA[cur][row * 40 + ka + 16];
        }
        ABFrag bfr[4];
        #pragma unroll
        for (int ni = 0; ni < 4; ++ni) {
            int col = wn * 64 + ni * 16 + l15;
            int kb2 = hb * 8;
            bfr[ni].h[0] = *(v8bf*)&sB[cur][col * 40 + kb2];
            bfr[ni].h[1] = *(v8bf*)&sB[cur][col * 40 + kb2 + 16];
        }
        #pragma unroll
        for (int mi = 0; mi < 4; ++mi)
            #pragma unroll
            for (int ni = 0; ni < 4; ++ni)
                acc[mi][ni] = wmma_bf16(afr[mi].v, bfr[ni].v, acc[mi][ni]);
        __syncthreads();   // protect buffer 'cur' from next iteration's fill
    }

    #pragma unroll
    for (int mi = 0; mi < 4; ++mi) {
        #pragma unroll
        for (int ni = 0; ni < 4; ++ni) {
            #pragma unroll
            for (int r = 0; r < 8; ++r) {
                int row = m0 + wm * 64 + mi * 16 + r + hb * 8;
                int col = n0 + wn * 64 + ni * 16 + l15;
                if (OUT_BF16)
                    ((__bf16*)Cout)[(size_t)row * N + col] = f2bf(acc[mi][ni][r]);
                else
                    ((float*)Cout)[(size_t)row * N + col] = acc[mi][ni][r];
            }
        }
    }
}

// ---------------- RoPE on Q: [B,T,D] f32 -> [B,H,T,DH] bf16 ----------------
__global__ __launch_bounds__(256) void rope_q_kernel(const float* __restrict__ qf,
                                                     const float* __restrict__ cosT,
                                                     const float* __restrict__ sinT,
                                                     __bf16* __restrict__ qo) {
    int idx = blockIdx.x * 256 + threadIdx.x;     // < B*T*D/2
    int p  = idx & 1023;
    int bt = idx >> 10;
    int hh = p >> 6;
    int i  = p & 63;
    int t  = bt & (Tz - 1);
    int b  = bt >> 11;
    const float* row = qf + (size_t)bt * Dz + hh * DHz;
    float x1 = row[2 * i], x2 = row[2 * i + 1];
    float c = cosT[t * 64 + i], s = sinT[t * 64 + i];
    __bf16* o = qo + (((size_t)(b * Hz + hh) * Tz) + t) * DHz;
    o[2 * i]     = f2bf(x1 * c - x2 * s);
    o[2 * i + 1] = f2bf(x1 * s + x2 * c);
}

// ---------------- RoPE on K + split V: [B,T,2D] f32 -> two [B,H,T,DH] bf16 ----------------
__global__ __launch_bounds__(256) void rope_kv_kernel(const float* __restrict__ kvf,
                                                      const float* __restrict__ cosT,
                                                      const float* __restrict__ sinT,
                                                      __bf16* __restrict__ ko,
                                                      __bf16* __restrict__ vo) {
    int idx = blockIdx.x * 256 + threadIdx.x;     // < B*T*1024
    int p  = idx & 1023;
    int bt = idx >> 10;
    int hh = p >> 6;
    int i  = p & 63;
    int t  = bt & (Tz - 1);
    int b  = bt >> 11;
    const float* row = kvf + (size_t)bt * (2 * Dz);
    float x1 = row[hh * DHz + 2 * i], x2 = row[hh * DHz + 2 * i + 1];
    float c = cosT[t * 64 + i], s = sinT[t * 64 + i];
    size_t ob = (((size_t)(b * Hz + hh) * Tz) + t) * DHz;
    ko[ob + 2 * i]     = f2bf(x1 * c - x2 * s);
    ko[ob + 2 * i + 1] = f2bf(x1 * s + x2 * c);
    float v1 = row[Dz + hh * DHz + 2 * i], v2 = row[Dz + hh * DHz + 2 * i + 1];
    vo[ob + 2 * i]     = f2bf(v1);
    vo[ob + 2 * i + 1] = f2bf(v2);
}

// ---------------- Flash attention (causal) ----------------
// grid (T/64, H, B), 128 threads = 4 waves; wave w owns query rows [w*16, w*16+16).
// Q/K/V layout: [B,H,T,128] bf16. Output written as [B,T,H*128] bf16.
#define KSTR 136
#define VSTR 72
#define PSTR 72
__global__ __launch_bounds__(128) void flash_attn_kernel(const __bf16* __restrict__ Q,
                                                         const __bf16* __restrict__ Kk,
                                                         const __bf16* __restrict__ Vv,
                                                         __bf16* __restrict__ Oo,
                                                         float scale) {
    __shared__ __bf16 sK[64 * KSTR];
    __shared__ __bf16 sVT[128 * VSTR];
    __shared__ __bf16 sP[4 * 16 * PSTR];

    const int tid  = threadIdx.x;
    const int lane = tid & 31;
    const int wv   = tid >> 5;
    const int hb   = lane >> 4;
    const int l15  = lane & 15;
    const int b  = blockIdx.z;
    const int hh = blockIdx.y;
    const int qt = blockIdx.x;

    const size_t headBase = ((size_t)(b * Hz + hh) * Tz) * DHz;
    const __bf16* qptr = Q + headBase;

    // Resident Q fragments: 16 rows x 128 dh as 4 K-step A-fragments
    ABFrag qa[4];
    {
        int qrow = qt * 64 + wv * 16 + l15;
        #pragma unroll
        for (int kk = 0; kk < 4; ++kk) {
            int d0 = kk * 32 + hb * 8;
            qa[kk].h[0] = *(const v8bf*)(qptr + (size_t)qrow * DHz + d0);
            qa[kk].h[1] = *(const v8bf*)(qptr + (size_t)qrow * DHz + d0 + 16);
        }
    }

    v8f o[8];
    #pragma unroll
    for (int nd = 0; nd < 8; ++nd) o[nd] = (v8f)(0.0f);
    float mrow[8], lrow[8];
    #pragma unroll
    for (int r = 0; r < 8; ++r) { mrow[r] = -1e30f; lrow[r] = 0.f; }

    const int ktiles = qt + 1;
    for (int kt = 0; kt < ktiles; ++kt) {
        __syncthreads();
        // K tile 64x128 -> sK via async global->LDS (row-major, padded stride)
        const __bf16* kp = Kk + headBase + (size_t)kt * 64 * DHz;
        #pragma unroll
        for (int i = tid; i < 64 * 16; i += 128) {
            int row = i >> 4, c = (i & 15) * 8;
            async_load_b128(lds_off32(&sK[row * KSTR + c]), kp + (size_t)row * DHz + c);
        }
        // V tile 64x128 transposed into sVT[dh][key] (manual: needs transpose)
        const __bf16* vp = Vv + headBase + (size_t)kt * 64 * DHz;
        for (int i = tid; i < 64 * 16; i += 128) {
            int row = i >> 4, c = (i & 15) * 8;
            v8bf x = *(const v8bf*)(vp + (size_t)row * DHz + c);
            #pragma unroll
            for (int e = 0; e < 8; ++e) sVT[(c + e) * VSTR + row] = x[e];
        }
        // Prefetch next tiles into cache while we compute this one.
        if (kt + 1 < ktiles) {
            __builtin_prefetch(kp + 64 * DHz + tid * 64, 0, 3);
            __builtin_prefetch(vp + 64 * DHz + tid * 64, 0, 3);
        }
        asm volatile("s_wait_asynccnt 0x0" ::: "memory");
        __syncthreads();

        // S = Q K^T  (16 rows x 64 keys per wave)
        v8f sacc[4];
        #pragma unroll
        for (int ni = 0; ni < 4; ++ni) sacc[ni] = (v8f)(0.0f);
        #pragma unroll
        for (int ni = 0; ni < 4; ++ni) {
            int key = ni * 16 + l15;
            #pragma unroll
            for (int kk = 0; kk < 4; ++kk) {
                ABFrag kb;
                int d0 = kk * 32 + hb * 8;
                kb.h[0] = *(v8bf*)&sK[key * KSTR + d0];
                kb.h[1] = *(v8bf*)&sK[key * KSTR + d0 + 16];
                sacc[ni] = wmma_bf16(qa[kk].v, kb.v, sacc[ni]);
            }
        }

        const bool needmask = (kt == qt);
        // online softmax, per row r (this lane-half handles rows r + hb*8)
        #pragma unroll
        for (int r = 0; r < 8; ++r) {
            int qrow = qt * 64 + wv * 16 + r + hb * 8;
            float rowmax = -1e30f;
            #pragma unroll
            for (int ni = 0; ni < 4; ++ni) {
                float x = sacc[ni][r] * scale;
                if (needmask) {
                    int key = kt * 64 + ni * 16 + l15;
                    if (key > qrow) x = -1e30f;
                }
                sacc[ni][r] = x;
                rowmax = fmaxf(rowmax, x);
            }
            for (int off = 1; off < 16; off <<= 1)
                rowmax = fmaxf(rowmax, __shfl_xor(rowmax, off, 32));
            float mo = mrow[r];
            float mn = fmaxf(mo, rowmax);
            float alpha = __expf(mo - mn);
            float psum = 0.f;
            #pragma unroll
            for (int ni = 0; ni < 4; ++ni) {
                float pv = __expf(sacc[ni][r] - mn);
                sacc[ni][r] = pv;
                psum += pv;
            }
            for (int off = 1; off < 16; off <<= 1)
                psum += __shfl_xor(psum, off, 32);
            lrow[r] = lrow[r] * alpha + psum;
            mrow[r] = mn;
            #pragma unroll
            for (int nd = 0; nd < 8; ++nd) o[nd][r] *= alpha;
        }

        // stage P (bf16) into per-wave LDS region for A-fragment reload
        __bf16* pw = &sP[wv * 16 * PSTR];
        #pragma unroll
        for (int ni = 0; ni < 4; ++ni)
            #pragma unroll
            for (int r = 0; r < 8; ++r)
                pw[(r + hb * 8) * PSTR + ni * 16 + l15] = f2bf(sacc[ni][r]);

        // O += P * V
        #pragma unroll
        for (int kk = 0; kk < 2; ++kk) {
            ABFrag pa;
            int kb0 = kk * 32 + hb * 8;
            pa.h[0] = *(v8bf*)&pw[l15 * PSTR + kb0];
            pa.h[1] = *(v8bf*)&pw[l15 * PSTR + kb0 + 16];
            #pragma unroll
            for (int nd = 0; nd < 8; ++nd) {
                ABFrag vb;
                int dh = nd * 16 + l15;
                int kv0 = kk * 32 + hb * 8;
                vb.h[0] = *(v8bf*)&sVT[dh * VSTR + kv0];
                vb.h[1] = *(v8bf*)&sVT[dh * VSTR + kv0 + 16];
                o[nd] = wmma_bf16(pa.v, vb.v, o[nd]);
            }
        }
    }

    // normalize and store to [B,T,H*DH] bf16
    float inv[8];
    #pragma unroll
    for (int r = 0; r < 8; ++r) inv[r] = 1.0f / lrow[r];
    #pragma unroll
    for (int nd = 0; nd < 8; ++nd) {
        #pragma unroll
        for (int r = 0; r < 8; ++r) {
            int t = qt * 64 + wv * 16 + r + hb * 8;
            size_t oi = ((size_t)(b * Tz + t)) * Dz + hh * DHz + nd * 16 + l15;
            Oo[oi] = f2bf(o[nd][r] * inv[r]);
        }
    }
}

// ---------------- host launcher ----------------
extern "C" void kernel_launch(void* const* d_in, const int* in_sizes, int n_in,
                              void* d_out, int out_size, void* d_ws, size_t ws_size,
                              hipStream_t stream) {
    (void)in_sizes; (void)n_in; (void)out_size; (void)ws_size;
    const float* Hq     = (const float*)d_in[0];
    const float* hub    = (const float*)d_in[1];
    const float* Wq     = (const float*)d_in[2];
    const float* norm_w = (const float*)d_in[3];
    const float* Wdown  = (const float*)d_in[4];
    const float* Wup    = (const float*)d_in[5];
    const float* Wo     = (const float*)d_in[6];
    const float* cosT   = (const float*)d_in[7];
    const float* sinT   = (const float*)d_in[8];
    float* out = (float*)d_out;

    char* ws = (char*)d_ws;
    size_t cur = 0;
    auto alloc = [&](size_t bytes) -> void* {
        void* p = ws + cur;
        cur += (bytes + 255) & ~(size_t)255;
        return p;
    };

    const size_t MT = (size_t)Bz * Tz;                 // 8192 rows
    __bf16* hq_b   = (__bf16*)alloc(MT * Dz * 2);
    __bf16* wq_b   = (__bf16*)alloc((size_t)Dz * Dz * 2);
    __bf16* wdn_b  = (__bf16*)alloc((size_t)DLATz * DHUBz * 2);
    __bf16* wup_b  = (__bf16*)alloc((size_t)2 * Dz * DLATz * 2);
    __bf16* wo_b   = (__bf16*)alloc((size_t)Dz * Dz * 2);
    __bf16* hubn_b = (__bf16*)alloc(MT * DHUBz * 2);
    __bf16* ckv_b  = (__bf16*)alloc(MT * DLATz * 2);
    __bf16* q_b    = (__bf16*)alloc(MT * Dz * 2);      // [B,H,T,DH]
    __bf16* k_b    = (__bf16*)alloc(MT * Dz * 2);      // [B,H,T,DH]
    __bf16* v_b    = (__bf16*)alloc(MT * Dz * 2);      // [B,H,T,DH]
    __bf16* attn_b = (__bf16*)alloc(MT * Dz * 2);      // [B,T,D]
    float*  big_f  = (float*)alloc(MT * (size_t)(2 * Dz) * 4);  // q_f32 then kv_f32

    // 1) conversions
    auto cvt = [&](const float* s, __bf16* d, size_t n) {
        cvt_bf16_kernel<<<(unsigned)(n / 256), 256, 0, stream>>>(s, d, n);
    };
    cvt(Hq,    hq_b,  MT * Dz);
    cvt(Wq,    wq_b,  (size_t)Dz * Dz);
    cvt(Wdown, wdn_b, (size_t)DLATz * DHUBz);
    cvt(Wup,   wup_b, (size_t)2 * Dz * DLATz);
    cvt(Wo,    wo_b,  (size_t)Dz * Dz);

    // 2) RMSNorm -> bf16
    rmsnorm_kernel<<<(unsigned)MT, 256, 0, stream>>>(hub, norm_w, hubn_b);

    // 3) q = Hq * Wq^T  -> f32 (first half of big_f)
    gemm_bf16_kernel<false><<<dim3(Dz / 128, (unsigned)(MT / 256)), 256, 0, stream>>>(
        hq_b, wq_b, big_f, (int)MT, Dz, Dz);

    // 4) RoPE q -> [B,H,T,DH] bf16
    rope_q_kernel<<<(unsigned)(MT * Dz / 2 / 256), 256, 0, stream>>>(big_f, cosT, sinT, q_b);

    // 5) c_kv = hub_n * Wdown^T -> bf16
    gemm_bf16_kernel<true><<<dim3(DLATz / 128, (unsigned)(MT / 256)), 256, 0, stream>>>(
        hubn_b, wdn_b, ckv_b, (int)MT, DLATz, DHUBz);

    // 6) kv = c_kv * Wup^T -> f32 (reuses big_f; q_f32 already consumed)
    gemm_bf16_kernel<false><<<dim3((2 * Dz) / 128, (unsigned)(MT / 256)), 256, 0, stream>>>(
        ckv_b, wup_b, big_f, (int)MT, 2 * Dz, DLATz);

    // 7) RoPE k + split v -> [B,H,T,DH] bf16
    rope_kv_kernel<<<(unsigned)(MT * (Dz / 2) / 256), 256, 0, stream>>>(big_f, cosT, sinT, k_b, v_b);

    // 8) causal flash attention -> attn_b [B,T,D] bf16
    flash_attn_kernel<<<dim3(Tz / 64, Hz, Bz), 128, 0, stream>>>(
        q_b, k_b, v_b, attn_b, 1.0f / sqrtf((float)DHz));

    // 9) out = attn * Wo^T -> f32 d_out
    gemm_bf16_kernel<false><<<dim3(Dz / 128, (unsigned)(MT / 256)), 256, 0, stream>>>(
        attn_b, wo_b, out, (int)MT, Dz, Dz);
}
